// SimCLRLoss_33801392620027
// MI455X (gfx1250) — matrix operand order repaired and checked by reading, not compiled, not run on hardware
//
#include <hip/hip_runtime.h>
#include <hip/hip_bf16.h>

// ---------------- problem constants ----------------
#define B_ROWS 4096
#define DIM 512
#define TAU_INV 10.0f
#define CHUNKS 8
#define COLS_TOTAL (2 * B_ROWS)
#define COLS_PER_CHUNK (COLS_TOTAL / CHUNKS)   // 1024
#define ROWS_PER_WG 128
#define TILE_N 16
#define NTILES (COLS_PER_CHUNK / TILE_N)       // 64

// LDS: only the double-buffered key tiles. Additive pad (1024+16 B pitch)
// keeps the 16 key rows on distinct bank groups AND makes per-k fragment
// addresses pure immediates for ds_load offset folding.
#define ROW_PITCH 1040
#define BTILE_BYTES (TILE_N * ROW_PITCH)       // 16640
#define SMEM_BYTES (2 * BTILE_BYTES)           // 33280

// workspace byte offsets
#define WS_AN 0u
#define WS_BN 4194304u
#define WS_TGT 8388608u
#define WS_SPART 8404992u
#define WS_BVAL 8667136u
#define WS_BIDX 8798208u

typedef __attribute__((ext_vector_type(16))) __bf16 v16bf;
typedef __attribute__((ext_vector_type(8)))  __bf16 v8bf;
typedef __attribute__((ext_vector_type(8)))  float  v8f;

// -------- CDNA5 async memory->LDS copy (ASYNCcnt-tracked, no data VGPRs) ----
__device__ __forceinline__ void async_copy_b128(const void* gptr, unsigned lds_off) {
  // VDST = per-lane LDS byte offset, VADDR = 64-bit global address, no SADDR.
  asm volatile("global_load_async_to_lds_b128 %0, %1, off"
               :: "v"(lds_off), "v"(gptr)
               : "memory");
}
__device__ __forceinline__ void wait_async0() {
#if __has_builtin(__builtin_amdgcn_s_wait_asynccnt)
  __builtin_amdgcn_s_wait_asynccnt(0);
#else
  asm volatile("s_wait_asynccnt 0x0" ::: "memory");
#endif
}

// ---------------------------------------------------------------------------
// Kernel 1: L2-normalize rows of a and b into bf16, and compute the exact
// fp32 target logit dot(a_i, b_i) * 10 per row. One wave32 per row.
// ---------------------------------------------------------------------------
__global__ __launch_bounds__(256) void simclr_norm_kernel(
    const float* __restrict__ a, const float* __restrict__ b,
    __hip_bfloat16* __restrict__ an, __hip_bfloat16* __restrict__ bn,
    float* __restrict__ target) {
  const int wave = threadIdx.x >> 5;
  const int lane = threadIdx.x & 31;
  const int row = blockIdx.x * 8 + wave;

  const float2* a2 = (const float2*)(a + (size_t)row * DIM);
  const float2* b2 = (const float2*)(b + (size_t)row * DIM);

  float sa = 0.f, sb = 0.f, dab = 0.f;
  float2 av[8], bv[8];
#pragma unroll
  for (int u = 0; u < 8; ++u) {
    float2 x = a2[lane + u * 32];
    float2 y = b2[lane + u * 32];
    av[u] = x; bv[u] = y;
    sa  += x.x * x.x + x.y * x.y;
    sb  += y.x * y.x + y.y * y.y;
    dab += x.x * y.x + x.y * y.y;
  }
#pragma unroll
  for (int m = 16; m >= 1; m >>= 1) {
    sa  += __shfl_xor(sa, m, 32);
    sb  += __shfl_xor(sb, m, 32);
    dab += __shfl_xor(dab, m, 32);
  }
  const float ina = 1.0f / fmaxf(sqrtf(sa), 1e-12f);
  const float inb = 1.0f / fmaxf(sqrtf(sb), 1e-12f);

#pragma unroll
  for (int u = 0; u < 8; ++u) {
    const size_t pos = (size_t)row * DIM + (size_t)(lane + u * 32) * 2;
    an[pos]     = __float2bfloat16(av[u].x * ina);
    an[pos + 1] = __float2bfloat16(av[u].y * ina);
    bn[pos]     = __float2bfloat16(bv[u].x * inb);
    bn[pos + 1] = __float2bfloat16(bv[u].y * inb);
  }
  if (lane == 0) target[row] = dab * ina * inb * TAU_INV;
}

// ---------------------------------------------------------------------------
// Kernel 2: streaming-softmax over logits via v_wmma_f32_16x16x32_bf16.
// grid = (row blocks, column chunks, pass). pass 0: queries=a, keys=[b;a]
// (diag-masked second half). pass 1: queries=b, keys=[a;b].
// Query fragments live persistently in 128 VGPRs; key tiles are staged with
// global_load_async_to_lds_b128 into a 33KB double buffer.
// ---------------------------------------------------------------------------
__global__ __launch_bounds__(256) void simclr_logits_kernel(
    const __hip_bfloat16* __restrict__ an, const __hip_bfloat16* __restrict__ bn,
    float* __restrict__ s_part, float* __restrict__ bval_part,
    int* __restrict__ bidx_part) {
  extern __shared__ char smem[];
  const int tid  = threadIdx.x;
  const int wave = tid >> 5;
  const int lane = tid & 31;
  const int r0   = blockIdx.x * ROWS_PER_WG;
  const int cy   = blockIdx.y;
  const int pass = blockIdx.z;

  const __hip_bfloat16* Q = pass ? bn : an;
  const bool secondHalf = (cy >= CHUNKS / 2);
  const __hip_bfloat16* Keys = secondHalf ? Q : (pass ? an : bn);

  const int col0   = cy * COLS_PER_CHUNK;
  const int kbase0 = secondHalf ? (col0 - B_ROWS) : col0;

  const unsigned sbase = (unsigned)(size_t)(void*)smem; // wave-relative LDS off
  const int row16 = lane & 15;
  const int half  = lane >> 4;

  // ---- stage key tile 0 via async DMA ----
  {
#pragma unroll
    for (int i = 0; i < 4; ++i) {
      const int u = tid + i * 256;
      const int row = u >> 6, c = u & 63;
      const void* g = (const char*)Keys + ((size_t)(kbase0 + row) * DIM + c * 8) * 2;
      async_copy_b128(g, sbase + row * ROW_PITCH + (unsigned)(c << 4));
    }
  }

  // ---- load this wave's 16 query rows into persistent A fragments ----
  // 16x32 bf16 A fragment, contiguous k (same order as B): lane -> row (lane&15); per k-step, two 16B
// chunks at element offsets k0+half*16 and k0+half*16+8.
  v16bf areg[16];
  {
    const char* Ag = (const char*)Q +
        ((size_t)(r0 + wave * TILE_N + row16) * DIM) * 2;
#pragma unroll
    for (int k = 0; k < 16; ++k) {
      const int off = k * 64 + half * 32;
      v8bf a0 = *(const v8bf*)(Ag + off);
      v8bf a1 = *(const v8bf*)(Ag + off + 16);
      areg[k] = __builtin_shufflevector(a0, a1, 0,1,2,3,4,5,6,7,
                                        8,9,10,11,12,13,14,15);
    }
  }

  wait_async0();
  __syncthreads();

  // diagonal tile index for this wave (mask lives in the second key half)
  int j_diag = -1;
  if (secondHalf) {
    const int d = (r0 + wave * TILE_N) - kbase0;
    if (d >= 0 && d < COLS_PER_CHUNK) j_diag = d >> 4;
  }

  float ssum[8];
  float bv[8];
  int   bi[8];
#pragma unroll
  for (int v = 0; v < 8; ++v) { ssum[v] = 0.f; bv[v] = -3.0e38f; bi[v] = -1; }

  const int diag_v = row16 - 8 * half; // masked when v == diag_v

  for (int j = 0; j < NTILES; ++j) {
    // Lane-invariant fragment base: per-k deltas are pure immediates (k*64).
    // 32x16 bf16 B fragment, contiguous k (same order as A): lane -> key (lane&15); 16 contiguous elements
    // at K = k*32 + half*16.
    const char* Bl = smem + (j & 1) * BTILE_BYTES + row16 * ROW_PITCH + half * 32;

    // async-prefetch next key tile straight into the other LDS buffer
    if (j + 1 < NTILES) {
      const int kb = kbase0 + (j + 1) * TILE_N;
      const unsigned obase = sbase + (unsigned)(((j + 1) & 1) * BTILE_BYTES);
#pragma unroll
      for (int i = 0; i < 4; ++i) {
        const int u = tid + i * 256;
        const int row = u >> 6, c = u & 63;
        const void* g = (const char*)Keys + ((size_t)(kb + row) * DIM + c * 8) * 2;
        async_copy_b128(g, obase + row * ROW_PITCH + (unsigned)(c << 4));
      }
    }

    // ---- 16x16 logit tile over K=512; two independent fragment sets per
    // unrolled body so ds_loads for step k+1 are in flight during wmma k ----
    v8f acc = {};
#pragma unroll
    for (int k = 0; k < 16; k += 2) {
      v8bf p0 = *(const v8bf*)(Bl + k * 64);
      v8bf p1 = *(const v8bf*)(Bl + k * 64 + 16);
      v8bf p2 = *(const v8bf*)(Bl + k * 64 + 64);
      v8bf p3 = *(const v8bf*)(Bl + k * 64 + 80);
      v16bf bf0 = __builtin_shufflevector(p0, p1, 0,1,2,3,4,5,6,7,
                                          8,9,10,11,12,13,14,15);
      v16bf bf1 = __builtin_shufflevector(p2, p3, 0,1,2,3,4,5,6,7,
                                          8,9,10,11,12,13,14,15);
      acc = __builtin_amdgcn_wmma_f32_16x16x32_bf16(
          false, areg[k], false, bf0, (short)0, acc, false, false);
      acc = __builtin_amdgcn_wmma_f32_16x16x32_bf16(
          false, areg[k + 1], false, bf1, (short)0, acc, false, false);
    }

    const int colidx = col0 + j * TILE_N + row16;
    if (j == j_diag) {               // wave-uniform: at most one tile/chunk
#pragma unroll
      for (int v = 0; v < 8; ++v) {
        float logit = acc[v] * TAU_INV;
        if (diag_v == v) logit = -1.0e30f;             // diagonal mask
        ssum[v] += __expf(logit - 10.0f);
        if (logit > bv[v]) { bv[v] = logit; bi[v] = colidx; }
      }
    } else {
#pragma unroll
      for (int v = 0; v < 8; ++v) {
        float logit = acc[v] * TAU_INV;
        ssum[v] += __expf(logit - 10.0f);              // fixed shift: |logit|<=10
        if (logit > bv[v]) { bv[v] = logit; bi[v] = colidx; }
      }
    }

    wait_async0();     // next tile fully landed in LDS
    __syncthreads();
  }

  // ---- horizontal reduce across the 16-lane halves, write chunk partials ----
#pragma unroll
  for (int v = 0; v < 8; ++v) {
    float s = ssum[v];
    float m = bv[v];
    int   mi = bi[v];
#pragma unroll
    for (int x = 8; x >= 1; x >>= 1) {
      s += __shfl_xor(s, x, 16);
      float om = __shfl_xor(m, x, 16);
      int   oi = __shfl_xor(mi, x, 16);
      if (om > m || (om == m && oi < mi)) { m = om; mi = oi; }
    }
    if ((lane & 15) == 0) {
      const int row = r0 + wave * TILE_N + v + (half ? 8 : 0);
      s_part[((size_t)pass * B_ROWS + row) * CHUNKS + cy] = s;
      if (pass == 0) {
        bval_part[(size_t)row * CHUNKS + cy] = m;
        bidx_part[(size_t)row * CHUNKS + cy] = mi;
      }
    }
  }
}

// ---------------------------------------------------------------------------
// Kernel 3: combine chunk partials -> loss and accuracy (single block).
// ---------------------------------------------------------------------------
__global__ __launch_bounds__(256) void simclr_reduce_kernel(
    const float* __restrict__ s_part, const float* __restrict__ bval_part,
    const int* __restrict__ bidx_part, const float* __restrict__ target,
    float* __restrict__ out) {
  const int tid = threadIdx.x;
  float lsum = 0.f;
  int correct = 0;
  for (int row = tid; row < B_ROWS; row += 256) {
    float sa = 0.f, sb = 0.f;
#pragma unroll
    for (int c = 0; c < CHUNKS; ++c) sa += s_part[(size_t)row * CHUNKS + c];
#pragma unroll
    for (int c = 0; c < CHUNKS; ++c)
      sb += s_part[((size_t)B_ROWS + row) * CHUNKS + c];
    const float t = target[row];
    lsum += (10.0f + logf(sa) - t) + (10.0f + logf(sb) - t);

    float bvv = -3.0e38f;
    int bii = 0x7fffffff;
#pragma unroll
    for (int c = 0; c < CHUNKS; ++c) {
      const float v = bval_part[(size_t)row * CHUNKS + c];
      const int   i = bidx_part[(size_t)row * CHUNKS + c];
      if (v > bvv || (v == bvv && i < bii)) { bvv = v; bii = i; }
    }
    correct += (bii == row) ? 1 : 0;
  }

  __shared__ float sred[256];
  __shared__ int   cred[256];
  sred[tid] = lsum;
  cred[tid] = correct;
  __syncthreads();
  for (int s = 128; s > 0; s >>= 1) {
    if (tid < s) { sred[tid] += sred[tid + s]; cred[tid] += cred[tid + s]; }
    __syncthreads();
  }
  if (tid == 0) {
    out[0] = sred[0] / (2.0f * B_ROWS);
    out[1] = 100.0f * (float)cred[0] / (float)B_ROWS;
  }
}

// ---------------------------------------------------------------------------
extern "C" void kernel_launch(void* const* d_in, const int* in_sizes, int n_in,
                              void* d_out, int out_size, void* d_ws, size_t ws_size,
                              hipStream_t stream) {
  (void)in_sizes; (void)n_in; (void)out_size; (void)ws_size;
  const float* va = (const float*)d_in[0];
  const float* vb = (const float*)d_in[1];
  char* ws = (char*)d_ws;
  __hip_bfloat16* an = (__hip_bfloat16*)(ws + WS_AN);
  __hip_bfloat16* bn = (__hip_bfloat16*)(ws + WS_BN);
  float* target = (float*)(ws + WS_TGT);
  float* s_part = (float*)(ws + WS_SPART);
  float* bvalp  = (float*)(ws + WS_BVAL);
  int*   bidxp  = (int*)(ws + WS_BIDX);
  float* out = (float*)d_out;

  simclr_norm_kernel<<<B_ROWS / 8, 256, 0, stream>>>(va, vb, an, bn, target);
  simclr_logits_kernel<<<dim3(B_ROWS / ROWS_PER_WG, CHUNKS, 2), 256, SMEM_BYTES,
                         stream>>>(an, bn, s_part, bvalp, bidxp);
  simclr_reduce_kernel<<<1, 256, 0, stream>>>(s_part, bvalp, bidxp, target, out);
}
